// Task5GAT_84670985273816
// MI455X (gfx1250) — compile-verified, matrix operand order checked by reading, not run on hardware
//
#include <hip/hip_runtime.h>
#include <hip/hip_bf16.h>

// ---------------------------------------------------------------------------
// Types for CDNA5 WMMA
// ---------------------------------------------------------------------------
typedef __attribute__((ext_vector_type(16))) __bf16 v16bf;
typedef __attribute__((ext_vector_type(8)))  float  v8f;

// ---------------------------------------------------------------------------
// Helpers
// ---------------------------------------------------------------------------
__device__ __forceinline__ unsigned short f32_to_bf16_rne(float f) {
    unsigned int u = __float_as_uint(f);
    u += 0x7FFFu + ((u >> 16) & 1u);      // round-to-nearest-even
    return (unsigned short)(u >> 16);
}
__device__ __forceinline__ unsigned int pack_bf16x2(float lo, float hi) {
    return (unsigned int)f32_to_bf16_rne(lo) |
           ((unsigned int)f32_to_bf16_rne(hi) << 16);
}

// Order-preserving float -> uint mapping (for atomic max on floats)
__device__ __forceinline__ unsigned int fmap(float f) {
    unsigned int u = __float_as_uint(f);
    return (u & 0x80000000u) ? ~u : (u | 0x80000000u);
}
__device__ __forceinline__ float funmap(unsigned int u) {
    unsigned int b = (u & 0x80000000u) ? (u & 0x7FFFFFFFu) : ~u;
    return __uint_as_float(b);
}

__device__ __forceinline__ void atomic_add_f32(float* p, float v) {
    __hip_atomic_fetch_add(p, v, __ATOMIC_RELAXED, __HIP_MEMORY_SCOPE_AGENT);
}
__device__ __forceinline__ void atomic_max_u32(unsigned int* p, unsigned int v) {
    __hip_atomic_fetch_max(p, v, __ATOMIC_RELAXED, __HIP_MEMORY_SCOPE_AGENT);
}

// ---------------------------------------------------------------------------
// Zero-init kernel (graph-capture safe init of workspace)
// ---------------------------------------------------------------------------
__global__ void zero_words_kernel(unsigned int* p, size_t nwords) {
    size_t i = (size_t)blockIdx.x * blockDim.x + threadIdx.x;
    if (i < nwords) p[i] = 0u;
}

// ---------------------------------------------------------------------------
// WMMA GEMM: C[M, ldc] = A[M,K] (f32) x B[K,Nreal] (f32, zero-padded to NT*16)
//
// - 256-row M tile per block, 8 waves; each wave owns TWO 16-row sub-tiles
//   (reuses every B fragment twice -> 16 WMMAs per K-step per wave).
// - Double-buffered LDS (ping-pong): stage tile kt+32 while WMMA consumes
//   tile kt; single barrier per K-step.
// - f32 -> bf16 convert on the fly, packed 2-per-DWORD LDS stores.
// - f32 accumulate via v_wmma_f32_16x16x32_bf16.
// ---------------------------------------------------------------------------
template <int NT>
__global__ __launch_bounds__(256) void wmma_gemm_kernel(
    const float* __restrict__ A, const float* __restrict__ B,
    float* __restrict__ C, int M, int K, int Nreal, int ldc) {

    __shared__ unsigned short sA[2][256][32];      // [buf][m][k] bf16
    __shared__ unsigned short sB[2][NT * 16][32];  // [buf][n][k] bf16 (B^T)

    const int t     = threadIdx.x;
    const int wave  = t >> 5;
    const int lane  = t & 31;
    const int row16 = lane & 15;
    const int half  = lane >> 4;
    const int mBase = blockIdx.x * 256;

    v8f acc0[NT] = {};   // wave's sub-tile 0 (rows wave*32 + 0..15)
    v8f acc1[NT] = {};   // wave's sub-tile 1 (rows wave*32 + 16..31)

    // ---- staging helpers (coalesced global reads, packed bf16x2 DS stores)
    auto stageA = [&](int buf, int kt) {
        // 256 rows x 16 k-pairs; consecutive threads -> consecutive k in a row
        for (int i = t; i < 256 * 16; i += 256) {
            int r  = i >> 4;
            int k2 = (i & 15) << 1;
            int gr = mBase + r, gk = kt + k2;
            float v0 = 0.f, v1 = 0.f;
            if (gr < M) {
                const float* ap = A + (size_t)gr * K + gk;
                if (gk     < K) v0 = ap[0];
                if (gk + 1 < K) v1 = ap[1];
            }
            *(unsigned int*)&sA[buf][r][k2] = pack_bf16x2(v0, v1);
        }
    };
    auto stageB = [&](int buf, int kt) {
        // NT*16 cols x 16 k-pairs; consecutive threads -> consecutive n
        for (int i = t; i < NT * 16 * 16; i += 256) {
            int k2 = (i / (NT * 16)) << 1;
            int n  = i % (NT * 16);
            int gk = kt + k2;
            float v0 = 0.f, v1 = 0.f;
            if (n < Nreal) {
                if (gk     < K) v0 = B[(size_t)gk * Nreal + n];
                if (gk + 1 < K) v1 = B[(size_t)(gk + 1) * Nreal + n];
            }
            *(unsigned int*)&sB[buf][n][k2] = pack_bf16x2(v0, v1);
        }
    };

    // ---- prologue: stage first tile
    stageA(0, 0);
    stageB(0, 0);
    __syncthreads();

    for (int kt = 0; kt < K; kt += 32) {
        const int cur = (kt >> 5) & 1;
        const int nxt = cur ^ 1;

        if (kt + 32 < K) {
            // prefetch the tile after next (global_prefetch_b8)
            {
                int gr = mBase + (t >> 5) * 32 + (t & 31);
                int gk = kt + 64;
                if (gr < M && gk < K)
                    __builtin_prefetch(&A[(size_t)gr * K + gk], 0, 0);
            }
            stageA(nxt, kt + 32);   // overlaps with WMMA below
            stageB(nxt, kt + 32);
        }

        // ---- compute on current buffer
        v16bf a0 = *(const v16bf*)&sA[cur][wave * 32 + row16][half * 16];
        v16bf a1 = *(const v16bf*)&sA[cur][wave * 32 + 16 + row16][half * 16];
#pragma unroll
        for (int j = 0; j < NT; ++j) {
            v16bf b = *(const v16bf*)&sB[cur][j * 16 + row16][half * 16];
            acc0[j] = __builtin_amdgcn_wmma_f32_16x16x32_bf16(
                false, a0, false, b, (short)0, acc0[j], false, false);
            acc1[j] = __builtin_amdgcn_wmma_f32_16x16x32_bf16(
                false, a1, false, b, (short)0, acc1[j], false, false);
        }
        __syncthreads();   // next tile staged AND current reads done
    }

    // C/D layout: VGPR r -> M = r (lanes 0-15) / r+8 (lanes 16-31), N = lane&15
#pragma unroll
    for (int j = 0; j < NT; ++j) {
#pragma unroll
        for (int r = 0; r < 8; ++r) {
            int mr  = (half ? (r + 8) : r);
            int col = j * 16 + row16;
            int row0 = mBase + wave * 32 + mr;
            int row1 = row0 + 16;
            if (row0 < M) C[(size_t)row0 * ldc + col] = acc0[j][r];
            if (row1 < M) C[(size_t)row1 * ldc + col] = acc1[j][r];
        }
    }
}

// ---------------------------------------------------------------------------
// Per-(node,head) attention scores: a_s = <h[n,h,:], att_src[h,:]>, same a_d
// ---------------------------------------------------------------------------
__global__ void attn_scores_kernel(const float* __restrict__ h,
                                   const float* __restrict__ att_s,
                                   const float* __restrict__ att_d,
                                   float* __restrict__ a_s, float* __restrict__ a_d,
                                   int N, int H, int Cc, int ldh) {
    int idx = blockIdx.x * blockDim.x + threadIdx.x;
    if (idx >= N * H) return;
    int n = idx / H, hh = idx % H;
    const float* hp = h + (size_t)n * ldh + hh * Cc;
    float s = 0.f, d = 0.f;
    for (int c = 0; c < Cc; ++c) {
        float v = hp[c];
        s += v * att_s[hh * Cc + c];
        d += v * att_d[hh * Cc + c];
    }
    a_s[idx] = s;
    a_d[idx] = d;
}

// ---------------------------------------------------------------------------
// Edge pass 1: e = leaky_relu(a_s[src]+a_d[dst]); atomic segment-max per dst
// ---------------------------------------------------------------------------
template <int H>
__global__ void edge_logits_kernel(const long long* __restrict__ ei, int E, int Np,
                                   const float* __restrict__ a_s,
                                   const float* __restrict__ a_d,
                                   float* __restrict__ ebuf,
                                   unsigned int* __restrict__ mmax) {
    int e = blockIdx.x * blockDim.x + threadIdx.x;
    int Ep = E + Np;
    if (e >= Ep) return;
    long long s, d;
    if (e < E) { s = ei[e]; d = ei[E + e]; } else { s = d = (long long)(e - E); }
#pragma unroll
    for (int h = 0; h < H; ++h) {
        float v = a_s[s * H + h] + a_d[d * H + h];
        v = (v > 0.f) ? v : 0.2f * v;              // leaky_relu, slope 0.2
        ebuf[(size_t)e * H + h] = v;
        atomic_max_u32(&mmax[d * H + h], fmap(v));
    }
}

// ---------------------------------------------------------------------------
// Edge pass 2: ex = exp(e - m[dst]); atomic segment-sum per dst
// ---------------------------------------------------------------------------
template <int H>
__global__ void edge_exp_kernel(const long long* __restrict__ ei, int E, int Np,
                                const unsigned int* __restrict__ mmax,
                                float* __restrict__ ebuf, float* __restrict__ z) {
    int e = blockIdx.x * blockDim.x + threadIdx.x;
    int Ep = E + Np;
    if (e >= Ep) return;
    long long d;
    if (e < E) { d = ei[E + e]; } else { d = (long long)(e - E); }
#pragma unroll
    for (int h = 0; h < H; ++h) {
        float m  = funmap(mmax[d * H + h]);
        float ex = __expf(ebuf[(size_t)e * H + h] - m);
        ebuf[(size_t)e * H + h] = ex;
        atomic_add_f32(&z[d * H + h], ex);
    }
}

// ---------------------------------------------------------------------------
// Edge pass 3: agg[dst] += (ex/z[dst]) * h[src]   (one thread per edge*head)
// ---------------------------------------------------------------------------
template <int H, int C>
__global__ void edge_agg_kernel(const long long* __restrict__ ei, int E, int Np,
                                const float* __restrict__ ebuf,
                                const float* __restrict__ z,
                                const float* __restrict__ h, int ldh,
                                float* __restrict__ agg, int ldagg) {
    int gid = blockIdx.x * blockDim.x + threadIdx.x;
    int Ep = E + Np;
    if (gid >= Ep * H) return;
    int e = gid / H, hh = gid % H;
    long long s, d;
    if (e < E) { s = ei[e]; d = ei[E + e]; } else { s = d = (long long)(e - E); }
    float alpha = ebuf[(size_t)e * H + hh] / z[d * H + hh];
    const float* hp = h + (size_t)s * ldh + hh * C;
    float* ap = agg + (size_t)d * ldagg + hh * C;
#pragma unroll
    for (int c = 0; c < C; ++c) atomic_add_f32(&ap[c], alpha * hp[c]);
}

// ---------------------------------------------------------------------------
// h1 = elu(agg1 + b1)
// ---------------------------------------------------------------------------
__global__ void elu_bias_kernel(const float* __restrict__ agg,
                                const float* __restrict__ b,
                                float* __restrict__ out, int N, int F) {
    int idx = blockIdx.x * blockDim.x + threadIdx.x;
    if (idx >= N * F) return;
    float v = agg[idx] + b[idx % F];
    out[idx] = (v > 0.f) ? v : (__expf(v) - 1.0f);
}

// ---------------------------------------------------------------------------
// out = acc2 + b2  (heads=1 -> mean over heads is identity)
// ---------------------------------------------------------------------------
__global__ void final_bias_kernel(const float* __restrict__ acc,
                                  const float* __restrict__ b,
                                  float* __restrict__ out, int N, int F) {
    int idx = blockIdx.x * blockDim.x + threadIdx.x;
    if (idx >= N * F) return;
    out[idx] = acc[idx] + b[idx % F];
}

// ---------------------------------------------------------------------------
// Host launcher
// ---------------------------------------------------------------------------
extern "C" void kernel_launch(void* const* d_in, const int* in_sizes, int n_in,
                              void* d_out, int out_size, void* d_ws, size_t ws_size,
                              hipStream_t stream) {
    (void)n_in; (void)out_size; (void)ws_size;

    const float*     x   = (const float*)d_in[0];
    const long long* ei  = (const long long*)d_in[1];
    const float*     W1  = (const float*)d_in[2];
    const float*     as1 = (const float*)d_in[3];
    const float*     ad1 = (const float*)d_in[4];
    const float*     b1  = (const float*)d_in[5];
    const float*     W2  = (const float*)d_in[6];
    const float*     as2 = (const float*)d_in[7];
    const float*     ad2 = (const float*)d_in[8];
    const float*     b2  = (const float*)d_in[9];
    float*           out = (float*)d_out;

    const int FIN = 1433, HEADS = 8, HID = 16, NCLS = 7;
    const int F1 = HEADS * HID;                 // 128
    const int N  = in_sizes[0] / FIN;           // 50000
    const int E  = in_sizes[1] / 2;             // 1600000
    const int Ep = E + N;                       // edges incl. self loops

    // ---- carve workspace ----
    char* p = (char*)d_ws;
    auto carve = [&](size_t bytes) -> char* {
        char* r = p;
        p += (bytes + 255) & ~(size_t)255;
        return r;
    };
    float* h1raw = (float*)carve((size_t)N * F1 * 4);   // layer-1 projections
    float* ebuf1 = (float*)carve((size_t)Ep * HEADS * 4);
    float* h1act = (float*)carve((size_t)N * F1 * 4);   // elu(agg1 + b1)
    float* h2p   = (float*)carve((size_t)N * 16 * 4);   // layer-2 proj (padded to 16)
    float* a_s1  = (float*)carve((size_t)N * HEADS * 4);
    float* a_d1  = (float*)carve((size_t)N * HEADS * 4);
    float* a_s2  = (float*)carve((size_t)N * 4);
    float* a_d2  = (float*)carve((size_t)N * 4);
    float* ebuf2 = (float*)carve((size_t)Ep * 4);
    // zero-initialized region (one contiguous span)
    char* zbase = p;
    unsigned int* m1   = (unsigned int*)carve((size_t)N * HEADS * 4);
    float*        z1   = (float*)carve((size_t)N * HEADS * 4);
    float*        agg1 = (float*)carve((size_t)N * F1 * 4);
    unsigned int* m2   = (unsigned int*)carve((size_t)N * 4);
    float*        z2   = (float*)carve((size_t)N * 4);
    float*        acc2 = (float*)carve((size_t)N * NCLS * 4);
    size_t zwords = (size_t)(p - zbase) / 4;

    const int TB = 256;
    auto cdiv = [](long long a, long long b) { return (int)((a + b - 1) / b); };

    // 0) zero segment-reduction buffers
    zero_words_kernel<<<cdiv((long long)zwords, TB), TB, 0, stream>>>(
        (unsigned int*)zbase, zwords);

    // 1) layer-1 GEMM: h1raw[N,128] = x @ W1   (bf16 WMMA, f32 accumulate)
    wmma_gemm_kernel<8><<<cdiv(N, 256), TB, 0, stream>>>(x, W1, h1raw, N, FIN, F1, F1);

    // 2) attention scores [N,8]
    attn_scores_kernel<<<cdiv((long long)N * HEADS, TB), TB, 0, stream>>>(
        h1raw, as1, ad1, a_s1, a_d1, N, HEADS, HID, F1);

    // 3) edge softmax + aggregate (layer 1)
    edge_logits_kernel<8><<<cdiv(Ep, TB), TB, 0, stream>>>(ei, E, N, a_s1, a_d1, ebuf1, m1);
    edge_exp_kernel<8><<<cdiv(Ep, TB), TB, 0, stream>>>(ei, E, N, m1, ebuf1, z1);
    edge_agg_kernel<8, 16><<<cdiv((long long)Ep * HEADS, TB), TB, 0, stream>>>(
        ei, E, N, ebuf1, z1, h1raw, F1, agg1, F1);

    // 4) h1 = elu(agg1 + b1)
    elu_bias_kernel<<<cdiv((long long)N * F1, TB), TB, 0, stream>>>(agg1, b1, h1act, N, F1);

    // 5) layer-2 GEMM: h2p[N,16(pad)] = h1 @ W2 (Nreal = 7, zero-padded)
    wmma_gemm_kernel<1><<<cdiv(N, 256), TB, 0, stream>>>(h1act, W2, h2p, N, F1, NCLS, 16);

    // 6) layer-2 attention scores (heads = 1)
    attn_scores_kernel<<<cdiv(N, TB), TB, 0, stream>>>(h2p, as2, ad2, a_s2, a_d2, N, 1, NCLS, 16);

    // 7) edge softmax + aggregate (layer 2)
    edge_logits_kernel<1><<<cdiv(Ep, TB), TB, 0, stream>>>(ei, E, N, a_s2, a_d2, ebuf2, m2);
    edge_exp_kernel<1><<<cdiv(Ep, TB), TB, 0, stream>>>(ei, E, N, m2, ebuf2, z2);
    edge_agg_kernel<1, 7><<<cdiv(Ep, TB), TB, 0, stream>>>(
        ei, E, N, ebuf2, z2, h2p, 16, acc2, NCLS);

    // 8) out = acc2 + b2
    final_bias_kernel<<<cdiv((long long)N * NCLS, TB), TB, 0, stream>>>(acc2, b2, out, N, NCLS);
}
